// TransformerModel_37598143709434
// MI455X (gfx1250) — matrix operand order repaired and checked by reading, not compiled, not run on hardware
//
#include <hip/hip_runtime.h>
#include <hip/hip_bf16.h>

typedef __attribute__((ext_vector_type(16))) _Float16 v16h;
typedef __attribute__((ext_vector_type(8)))  float    v8f;

#define NND   50000
#define NE    800000
#define NH    4
#define HD    32
#define HIDW  128
#define QW    512     // q|k|v|skip concatenated
#define NG    64

// ---------------- utility fills ----------------
__global__ void fill_kernel(float* __restrict__ p, int n, float v) {
    int i = blockIdx.x * blockDim.x + threadIdx.x;
    if (i < n) p[i] = v;
}

// ---------------- weight prepack to WMMA B layout (f16) ----------------
// Packed index = ((kb*32 + cb)*32 + lane)*16 + i
// value = W_sel[kb*32 + (lane>>4)*8 + (i&7) + (i>>3)*16][col & 127], col = cb*16 + (lane&15)
__global__ void pack_weights(const float* __restrict__ Wq, const float* __restrict__ Wk,
                             const float* __restrict__ Wv, const float* __restrict__ Ws,
                             _Float16* __restrict__ Bp, int K) {
    int tid = blockIdx.x * blockDim.x + threadIdx.x;
    if (tid >= K * QW) return;
    int i    = tid & 15;
    int lane = (tid >> 4) & 31;
    int cb   = (tid >> 9) & 31;
    int kb   = tid >> 14;
    int col  = cb * 16 + (lane & 15);
    int krow = kb * 32 + (lane >> 4) * 8 + (i & 7) + (i >> 3) * 16;
    int sel  = col >> 7;
    int lc   = col & 127;
    const float* W = (sel == 0) ? Wq : (sel == 1) ? Wk : (sel == 2) ? Wv : Ws;
    Bp[tid] = (_Float16)W[krow * HIDW + lc];
}

__global__ void pack_bias(const float* __restrict__ bq, const float* __restrict__ bk,
                          const float* __restrict__ bv, const float* __restrict__ bs,
                          float* __restrict__ bias) {
    int tid = blockIdx.x * blockDim.x + threadIdx.x;
    if (tid >= QW) return;
    int sel = tid >> 7, j = tid & 127;
    const float* b = (sel == 0) ? bq : (sel == 1) ? bk : (sel == 2) ? bv : bs;
    bias[tid] = b[j];
}

// ---------------- fused q|k|v|s projection GEMM via WMMA ----------------
// Y[16 x 512] per block; 8 waves, each wave owns 4 col tiles (64 cols).
__global__ void __launch_bounds__(256) gemm_qkvs(const float* __restrict__ X,
                                                 const _Float16* __restrict__ Bp,
                                                 const float* __restrict__ bias,
                                                 float* __restrict__ Y, int K) {
    const int lane  = threadIdx.x & 31;
    const int wave  = threadIdx.x >> 5;
    const int m15   = lane & 15;
    const int khalf = lane >> 4;
    const int r0    = blockIdx.x * 16;
    const int row   = r0 + m15;

    v8f acc0 = {}, acc1 = {}, acc2 = {}, acc3 = {};
    const int nkb = K >> 5;
    const v16h* Bv = (const v16h*)Bp;

    for (int kb = 0; kb < nkb; ++kb) {
        // A fragment: 16-bit A 16x32 layout (lane<16: K 0..7 & 16..23; lane>=16: +8)
        const float* ap = X + (size_t)row * K + kb * 32 + khalf * 8;
        float4 f0 = *(const float4*)(ap);
        float4 f1 = *(const float4*)(ap + 4);
        float4 f2 = *(const float4*)(ap + 16);
        float4 f3 = *(const float4*)(ap + 20);
        v16h a;
        a[0]=(_Float16)f0.x;  a[1]=(_Float16)f0.y;  a[2]=(_Float16)f0.z;  a[3]=(_Float16)f0.w;
        a[4]=(_Float16)f1.x;  a[5]=(_Float16)f1.y;  a[6]=(_Float16)f1.z;  a[7]=(_Float16)f1.w;
        a[8]=(_Float16)f2.x;  a[9]=(_Float16)f2.y;  a[10]=(_Float16)f2.z; a[11]=(_Float16)f2.w;
        a[12]=(_Float16)f3.x; a[13]=(_Float16)f3.y; a[14]=(_Float16)f3.z; a[15]=(_Float16)f3.w;

        const v16h* bb = Bv + (((size_t)kb * 32 + wave * 4) * 32 + lane);
        v16h b0 = bb[0];
        v16h b1 = bb[32];
        v16h b2 = bb[64];
        v16h b3 = bb[96];
        acc0 = __builtin_amdgcn_wmma_f32_16x16x32_f16(false, a, false, b0, (short)0, acc0, false, false);
        acc1 = __builtin_amdgcn_wmma_f32_16x16x32_f16(false, a, false, b1, (short)0, acc1, false, false);
        acc2 = __builtin_amdgcn_wmma_f32_16x16x32_f16(false, a, false, b2, (short)0, acc2, false, false);
        acc3 = __builtin_amdgcn_wmma_f32_16x16x32_f16(false, a, false, b3, (short)0, acc3, false, false);
    }

    // C layout: vgpr j -> row r0 + khalf*8 + j, col cb*16 + m15
    const size_t rbase = (size_t)(r0 + khalf * 8) * QW;
#define STORE_TILE(ACC, T)                                                \
    {                                                                     \
        const int col = (wave * 4 + (T)) * 16 + m15;                      \
        const float bv = bias[col];                                       \
        float* yp = Y + rbase + col;                                      \
        yp[0*QW]=ACC[0]+bv; yp[1*QW]=ACC[1]+bv; yp[2*QW]=ACC[2]+bv;       \
        yp[3*QW]=ACC[3]+bv; yp[4*QW]=ACC[4]+bv; yp[5*QW]=ACC[5]+bv;       \
        yp[6*QW]=ACC[6]+bv; yp[7*QW]=ACC[7]+bv;                           \
    }
    STORE_TILE(acc0, 0)
    STORE_TILE(acc1, 1)
    STORE_TILE(acc2, 2)
    STORE_TILE(acc3, 3)
#undef STORE_TILE
}

// ---------------- edge attention ----------------
__device__ __forceinline__ void atomicMaxF(float* addr, float v) {
    if (v >= 0.0f) atomicMax((int*)addr, __float_as_int(v));
    else           atomicMin((unsigned int*)addr, __float_as_uint(v));
}

__global__ void edge_logits(const int* __restrict__ ei, const float* __restrict__ QKVS,
                            float* __restrict__ P, float* __restrict__ Mx) {
    int tid = blockIdx.x * blockDim.x + threadIdx.x;
    if (tid >= NE * NH) return;
    int e = tid >> 2, h = tid & 3;
    int src = ei[e], dst = ei[NE + e];
    const float4* qp = (const float4*)(QKVS + (size_t)dst * QW + h * HD);        // q block
    const float4* kp = (const float4*)(QKVS + (size_t)src * QW + HIDW + h * HD); // k block
    float dot = 0.0f;
#pragma unroll
    for (int i = 0; i < 8; ++i) {
        float4 q = qp[i], k = kp[i];
        dot = fmaf(q.x, k.x, dot); dot = fmaf(q.y, k.y, dot);
        dot = fmaf(q.z, k.z, dot); dot = fmaf(q.w, k.w, dot);
    }
    float lg = dot * 0.17677669529663687f;  // 1/sqrt(32)
    P[tid] = lg;
    atomicMaxF(&Mx[dst * NH + h], lg);
}

__global__ void edge_exp(const int* __restrict__ ei, float* __restrict__ P,
                         const float* __restrict__ Mx, float* __restrict__ Sx) {
    int tid = blockIdx.x * blockDim.x + threadIdx.x;
    if (tid >= NE * NH) return;
    int e = tid >> 2, h = tid & 3;
    int dst = ei[NE + e];
    float p = expf(P[tid] - Mx[dst * NH + h]);
    P[tid] = p;
    atomicAdd(&Sx[dst * NH + h], p);
}

__global__ void edge_scatter(const int* __restrict__ ei, const float* __restrict__ P,
                             const float* __restrict__ Sx, const float* __restrict__ QKVS,
                             float* __restrict__ Hb) {
    int tid = blockIdx.x * blockDim.x + threadIdx.x;
    if (tid >= NE * NH) return;
    int e = tid >> 2, h = tid & 3;
    int src = ei[e], dst = ei[NE + e];
    float alpha = P[tid] / Sx[dst * NH + h];
    const float4* vp = (const float4*)(QKVS + (size_t)src * QW + 2 * HIDW + h * HD); // v block
    float* op = Hb + (size_t)dst * HIDW + h * HD;
#pragma unroll
    for (int i = 0; i < 8; ++i) {
        float4 v = vp[i];
        atomicAdd(op + i * 4 + 0, alpha * v.x);
        atomicAdd(op + i * 4 + 1, alpha * v.y);
        atomicAdd(op + i * 4 + 2, alpha * v.z);
        atomicAdd(op + i * 4 + 3, alpha * v.w);
    }
}

// out = relu(agg + skip)
__global__ void finalize_layer(const float* __restrict__ QKVS, float* __restrict__ Hb) {
    int tid = blockIdx.x * blockDim.x + threadIdx.x;
    if (tid >= NND * HIDW) return;
    int n = tid >> 7, c = tid & 127;
    float v = Hb[tid] + QKVS[(size_t)n * QW + 3 * HIDW + c];
    Hb[tid] = v > 0.0f ? v : 0.0f;
}

// ---------------- pooling + readout ----------------
__global__ void pool_accum(const int* __restrict__ batch, const float* __restrict__ Hb,
                           float* __restrict__ pooled, float* __restrict__ counts) {
    int tid = blockIdx.x * blockDim.x + threadIdx.x;
    if (tid >= NND * HIDW) return;
    int n = tid >> 7, c = tid & 127;
    int g = batch[n];
    atomicAdd(&pooled[g * HIDW + c], Hb[tid]);
    if (c == 0) atomicAdd(&counts[g], 1.0f);
}

__global__ void readout(const float* __restrict__ pooled, const float* __restrict__ counts,
                        const float* __restrict__ Wl, const float* __restrict__ bl,
                        float* __restrict__ out) {
    int tid = threadIdx.x;
    if (tid >= NG * 2) return;
    int g = tid >> 1, o = tid & 1;
    float inv = 1.0f / fmaxf(counts[g], 1.0f);
    float acc = 0.0f;
    for (int c = 0; c < HIDW; ++c)
        acc = fmaf(pooled[g * HIDW + c] * inv, Wl[c * 2 + o], acc);
    out[g * 2 + o] = acc + bl[o];
}

// ---------------- host side ----------------
static void run_layer(const float* X, int K,
                      const float* Wq, const float* bq, const float* Wk, const float* bk,
                      const float* Wv, const float* bv, const float* Ws, const float* bs,
                      const int* ei, float* QKVS, float* P, float* Mx, float* Sx,
                      float* Hb, float* bias, _Float16* Bp, hipStream_t stream) {
    pack_weights<<<(K * QW + 255) / 256, 256, 0, stream>>>(Wq, Wk, Wv, Ws, Bp, K);
    pack_bias<<<2, 256, 0, stream>>>(bq, bk, bv, bs, bias);
    gemm_qkvs<<<NND / 16, 256, 0, stream>>>(X, Bp, bias, QKVS, K);
    fill_kernel<<<(NND * NH + 255) / 256, 256, 0, stream>>>(Mx, NND * NH, -3.402823466e38f);
    fill_kernel<<<(NND * NH + 255) / 256, 256, 0, stream>>>(Sx, NND * NH, 0.0f);
    fill_kernel<<<(NND * HIDW + 255) / 256, 256, 0, stream>>>(Hb, NND * HIDW, 0.0f);
    edge_logits<<<(NE * NH) / 256, 256, 0, stream>>>(ei, QKVS, P, Mx);
    edge_exp<<<(NE * NH) / 256, 256, 0, stream>>>(ei, P, Mx, Sx);
    edge_scatter<<<(NE * NH) / 256, 256, 0, stream>>>(ei, P, Sx, QKVS, Hb);
    finalize_layer<<<(NND * HIDW) / 256, 256, 0, stream>>>(QKVS, Hb);
}

extern "C" void kernel_launch(void* const* d_in, const int* in_sizes, int n_in,
                              void* d_out, int out_size, void* d_ws, size_t ws_size,
                              hipStream_t stream) {
    const float* x     = (const float*)d_in[0];
    const int*   ei    = (const int*)d_in[1];
    const int*   batch = (const int*)d_in[2];
    const float* Wq1 = (const float*)d_in[3],  *bq1 = (const float*)d_in[4];
    const float* Wk1 = (const float*)d_in[5],  *bk1 = (const float*)d_in[6];
    const float* Wv1 = (const float*)d_in[7],  *bv1 = (const float*)d_in[8];
    const float* Ws1 = (const float*)d_in[9],  *bs1 = (const float*)d_in[10];
    const float* Wq2 = (const float*)d_in[11], *bq2 = (const float*)d_in[12];
    const float* Wk2 = (const float*)d_in[13], *bk2 = (const float*)d_in[14];
    const float* Wv2 = (const float*)d_in[15], *bv2 = (const float*)d_in[16];
    const float* Ws2 = (const float*)d_in[17], *bs2 = (const float*)d_in[18];
    const float* Wl  = (const float*)d_in[19], *bl  = (const float*)d_in[20];
    float* out = (float*)d_out;

    float* ws = (float*)d_ws;
    float* QKVS   = ws;                        // N*512      = 25,600,000
    float* P      = ws + 25600000;             // E*H        =  3,200,000
    float* Mx     = ws + 28800000;             // N*H        =    200,000
    float* Sx     = ws + 29000000;             // N*H        =    200,000
    float* H1     = ws + 29200000;             // N*128      =  6,400,000
    float* H2     = ws + 35600000;             // N*128      =  6,400,000
    float* pooled = ws + 42000000;             // G*128      =      8,192
    float* counts = pooled + NG * HIDW;        // G          =         64
    float* bias   = counts + NG;               // 512
    _Float16* Bp  = (_Float16*)(ws + 42008832); // 128*512 halves (32B aligned)

    run_layer(x,  64,  Wq1, bq1, Wk1, bk1, Wv1, bv1, Ws1, bs1,
              ei, QKVS, P, Mx, Sx, H1, bias, Bp, stream);
    run_layer(H1, 128, Wq2, bq2, Wk2, bk2, Wv2, bv2, Ws2, bs2,
              ei, QKVS, P, Mx, Sx, H2, bias, Bp, stream);

    fill_kernel<<<(NG * HIDW + NG + 255) / 256, 256, 0, stream>>>(pooled, NG * HIDW + NG, 0.0f);
    pool_accum<<<(NND * HIDW) / 256, 256, 0, stream>>>(batch, H2, pooled, counts);
    readout<<<1, 128, 0, stream>>>(pooled, counts, Wl, bl, out);
}
